// PHMLinear_39573828665680
// MI455X (gfx1250) — matrix operand order repaired and checked by reading, not compiled, not run on hardware
//
#include <hip/hip_runtime.h>
#include <hip/hip_bf16.h>
#include <stdint.h>

// ---------------------------------------------------------------------------
// PHM linear  ==  y = x @ H + b,  H = sum_i kron(rule[i], W[i])
//   x: (8192, 2048) f32,  rule: (4,4,4) f32, W: (4,512,2048) f32, b: (8192)
//   y: (8192, 8192) f32
// Strategy: materialize H^T in bf16 once (cheap), convert x to bf16 once,
// then a WMMA bf16 GEMM (f32 accumulate) does the 275-GFLOP heavy lifting.
// Data movement: x+H^T (64MB bf16) live in the 192MB L2; only y streams out.
// ---------------------------------------------------------------------------

typedef __attribute__((ext_vector_type(16))) __bf16 v16bf;
typedef __attribute__((ext_vector_type(8)))  float  v8f;

#define TOKENS  8192
#define IN_F    2048
#define OUT_F   8192
#define PHM     4
#define IN_PER  (IN_F / PHM)    // 512
#define OUT_PER (OUT_F / PHM)   // 2048

// GEMM tiling
#define BM 128
#define BN 128
#define BK 64
#define LDT (BK + 8)            // LDS row stride in bf16 (144B -> conflict-free)
#define KTILES (IN_F / BK)      // 32
#define TILE_ELEMS (BM * LDT)   // per matrix per buffer
#define LDS_BYTES  (2 * 2 * TILE_ELEMS * (int)sizeof(uint16_t))   // 73728 B

// --- CDNA5 async global->LDS path (no data VGPRs, tracked by ASYNCcnt) ------
#if defined(__has_builtin)
#  if __has_builtin(__builtin_amdgcn_global_load_async_to_lds_b128)
#    define HAVE_ASYNC_LDS 1
#  endif
#  if __has_builtin(__builtin_amdgcn_s_wait_asynccnt)
#    define HAVE_WAIT_ASYNC_BUILTIN 1
#  endif
#endif

typedef int v4i __attribute__((vector_size(16)));
typedef __attribute__((address_space(1))) v4i g_v4i;
typedef __attribute__((address_space(3))) v4i l_v4i;

__device__ __forceinline__ void copy16_to_lds(const uint16_t* g, uint16_t* l) {
#if defined(HAVE_ASYNC_LDS)
    v4i* gv = (v4i*)g;   // drop const, retype (generic AS)
    v4i* lv = (v4i*)l;
    __builtin_amdgcn_global_load_async_to_lds_b128((g_v4i*)gv, (l_v4i*)lv,
                                                   /*offset=*/0, /*cpol=*/0);
#else
    *(uint4*)l = *(const uint4*)g;   // fallback: global->VGPR->ds_store
#endif
}

__device__ __forceinline__ void wait_lds_copies() {
#if defined(HAVE_ASYNC_LDS)
#  if defined(HAVE_WAIT_ASYNC_BUILTIN)
    __builtin_amdgcn_s_wait_asynccnt(0);
#  else
    asm volatile("s_wait_asynccnt 0" ::: "memory");
#  endif
#endif
}

__device__ __forceinline__ uint32_t f32_to_bf16_rne(float f) {
    union { float f; uint32_t u; } v; v.f = f;
    uint32_t u = v.u;
    u += 0x7FFFu + ((u >> 16) & 1u);   // round-to-nearest-even
    return u >> 16;
}
__device__ __forceinline__ uint32_t pack_bf16x2(float lo, float hi) {
    return f32_to_bf16_rne(lo) | (f32_to_bf16_rne(hi) << 16);
}

// ---------------------------------------------------------------------------
// Kernel 1: x (f32) -> xb (bf16), 4 elems / thread
// ---------------------------------------------------------------------------
__global__ __launch_bounds__(256) void cvt_x_bf16(const float* __restrict__ x,
                                                  uint32_t* __restrict__ xb2) {
    int i = blockIdx.x * blockDim.x + threadIdx.x;   // float4 index
    float4 f = ((const float4*)x)[i];
    uint2 p;
    p.x = pack_bf16x2(f.x, f.y);
    p.y = pack_bf16x2(f.z, f.w);
    ((uint2*)xb2)[i] = p;
}

// ---------------------------------------------------------------------------
// Kernel 2: build H^T in bf16.  Ht[n][k] = H[k][n] = sum_i rule[i][a][t]*W[i][c][p]
//   k = a*IN_PER + c  (GEMM K dim),  n = t*OUT_PER + p (GEMM N dim)
// ---------------------------------------------------------------------------
__global__ __launch_bounds__(256) void build_ht(const float* __restrict__ rule,
                                                const float* __restrict__ W,
                                                uint32_t* __restrict__ ht2) {
    int idx = blockIdx.x * blockDim.x + threadIdx.x;       // pair index
    int k2  = (idx & (IN_F / 2 - 1)) * 2;                  // even k
    int n   = idx >> 10;                                   // idx / (IN_F/2)
    int t = n / OUT_PER, p = n % OUT_PER;
    int a = k2 / IN_PER, c = k2 % IN_PER;                  // same a for k2, k2+1
    float acc0 = 0.f, acc1 = 0.f;
#pragma unroll
    for (int i = 0; i < PHM; ++i) {
        float r  = rule[(i * PHM + a) * PHM + t];          // wave-uniform
        float w0 = W[(size_t)(i * IN_PER + c)     * OUT_PER + p];
        float w1 = W[(size_t)(i * IN_PER + c + 1) * OUT_PER + p];
        acc0 += r * w0;
        acc1 += r * w1;
    }
    ht2[idx] = pack_bf16x2(acc0, acc1);
}

// ---------------------------------------------------------------------------
// Kernel 3: C[M][N] = A[M][K] * Bt[N][K]^T + bias,  bf16 in, f32 out.
// 256 threads = 8 wave32 (2 x 4).  Wave tile 64x32 = 4x2 WMMA accumulators.
// Double-buffered LDS fed by async global->LDS copies.
// ---------------------------------------------------------------------------
union FragU { uint4 q[2]; v16bf v; };

__global__ __launch_bounds__(256) void phm_gemm_bf16(const uint16_t* __restrict__ A,
                                                     const uint16_t* __restrict__ Bt,
                                                     const float* __restrict__ bias,
                                                     float* __restrict__ C) {
    extern __shared__ __align__(16) uint16_t smem[];
    // layout: [buf0: A(128x72) | B(128x72)][buf1: A | B]

    const int tid  = threadIdx.x;
    const int lane = tid & 31;
    const int wave = tid >> 5;
    const int wm   = wave >> 2;         // 0..1 : 64-row slab
    const int wn   = wave & 3;          // 0..3 : 32-col slab
    const int half = lane >> 4;         // K-half selector per WMMA layout
    const int lr   = lane & 15;

    const int m0 = blockIdx.y * BM;
    const int n0 = blockIdx.x * BN;

    // Tile copy indexing: 128 rows x 8 quads/row = 1024 quads, 4 per thread:
    //   quad j: row = (tid>>3) + j*32, colquad = tid&7
    const int lrow = tid >> 3;
    const int lcv  = tid & 7;
    const uint16_t* gA = A  + (size_t)(m0 + lrow) * IN_F + lcv * 8;
    const uint16_t* gB = Bt + (size_t)(n0 + lrow) * IN_F + lcv * 8;
    const int loff = lrow * LDT + lcv * 8;

    const v8f vzero = {0.f, 0.f, 0.f, 0.f, 0.f, 0.f, 0.f, 0.f};
    v8f acc[4][2];
#pragma unroll
    for (int tm = 0; tm < 4; ++tm)
#pragma unroll
        for (int tn = 0; tn < 2; ++tn) acc[tm][tn] = vzero;

    // ---- prologue: tile 0 into buffer 0 ----
#pragma unroll
    for (int j = 0; j < 4; ++j) {
        copy16_to_lds(gA + (size_t)j * 32 * IN_F, smem + loff + j * 32 * LDT);
        copy16_to_lds(gB + (size_t)j * 32 * IN_F, smem + TILE_ELEMS + loff + j * 32 * LDT);
    }
    wait_lds_copies();
    __syncthreads();

#pragma clang loop unroll(disable)
    for (int kt = 0; kt < KTILES; ++kt) {
        const int cur = kt & 1;
        const int nxt = cur ^ 1;
        uint16_t* bufc = smem + cur * 2 * TILE_ELEMS;
        uint16_t* bufn = smem + nxt * 2 * TILE_ELEMS;

        // ---- kick async copies for tile kt+1 (no data VGPRs consumed) ----
        if (kt + 1 < KTILES) {
            const size_t koff = (size_t)(kt + 1) * BK;
#pragma unroll
            for (int j = 0; j < 4; ++j) {
                copy16_to_lds(gA + koff + (size_t)j * 32 * IN_F,
                              bufn + loff + j * 32 * LDT);
                copy16_to_lds(gB + koff + (size_t)j * 32 * IN_F,
                              bufn + TILE_ELEMS + loff + j * 32 * LDT);
            }
        }

        // ---- compute on current tile: 2 k-steps of 32 ----
        const uint16_t* sA = bufc;
        const uint16_t* sB = bufc + TILE_ELEMS;
#pragma unroll
        for (int ks = 0; ks < BK; ks += 32) {
            // B fragments for this k-step stay live (16 VGPRs)...
            FragU bf0, bf1;
            {
                const uint16_t* p = sB + (wn * 32 + lr) * LDT + ks + half * 16;
                bf0.q[0] = *(const uint4*)(p);
                bf0.q[1] = *(const uint4*)(p + 8);
                p += 16 * LDT;
                bf1.q[0] = *(const uint4*)(p);
                bf1.q[1] = *(const uint4*)(p + 8);
            }
            // ...while A fragments stream through one 8-VGPR tuple.
#pragma unroll
            for (int tm = 0; tm < 4; ++tm) {
                FragU af;
                const uint16_t* p = sA + (wm * 64 + tm * 16 + lr) * LDT + ks + half * 16;
                af.q[0] = *(const uint4*)(p);        // elems 0..7 : K +half*16
                af.q[1] = *(const uint4*)(p + 8);    // elems 8..15: K +half*16+8
                acc[tm][0] = __builtin_amdgcn_wmma_f32_16x16x32_bf16(
                    false, af.v, false, bf0.v, (short)0, acc[tm][0], false, false);
                acc[tm][1] = __builtin_amdgcn_wmma_f32_16x16x32_bf16(
                    false, af.v, false, bf1.v, (short)0, acc[tm][1], false, false);
            }
        }

        // ---- drain this wave's async copies, then block-wide handoff ----
        wait_lds_copies();
        __syncthreads();
    }

    // ---- epilogue: bias + store.  D layout: VGPR r -> M = r + half*8, N = lr ----
#pragma unroll
    for (int tn = 0; tn < 2; ++tn) {
        const int col = n0 + wn * 32 + tn * 16 + lr;
        const float bv = bias[col];
#pragma unroll
        for (int tm = 0; tm < 4; ++tm) {
            const int rowb = m0 + wm * 64 + tm * 16 + half * 8;
#pragma unroll
            for (int r = 0; r < 8; ++r)
                C[(size_t)(rowb + r) * OUT_F + col] = acc[tm][tn][r] + bv;
        }
    }
}

// ---------------------------------------------------------------------------
extern "C" void kernel_launch(void* const* d_in, const int* in_sizes, int n_in,
                              void* d_out, int out_size, void* d_ws, size_t ws_size,
                              hipStream_t stream) {
    const float* x    = (const float*)d_in[0];
    const float* rule = (const float*)d_in[1];
    const float* W    = (const float*)d_in[2];
    const float* bias = (const float*)d_in[3];
    float* out = (float*)d_out;

    // workspace: [ xb : 8192*2048 bf16 = 32MB | ht : 8192*2048 bf16 = 32MB ]
    uint16_t* xb = (uint16_t*)d_ws;
    uint16_t* ht = (uint16_t*)((char*)d_ws + (size_t)TOKENS * IN_F * sizeof(uint16_t));

    cvt_x_bf16<<<(TOKENS * IN_F / 4) / 256, 256, 0, stream>>>(x, (uint32_t*)xb);
    build_ht<<<(OUT_F * (IN_F / 2)) / 256, 256, 0, stream>>>(rule, W, (uint32_t*)ht);

    dim3 grid(OUT_F / BN, TOKENS / BM);   // (64, 64)
    phm_gemm_bf16<<<grid, 256, LDS_BYTES, stream>>>(xb, ht, bias, out);
}
